// SO8TPhi3Model_87084756894256
// MI455X (gfx1250) — matrix operand order repaired and checked by reading, not compiled, not run on hardware
//
#include <hip/hip_runtime.h>
#include <math.h>

// ---------------------------------------------------------------------------
// SO8T-Phi pipeline for MI455X (gfx1250, wave32, WMMA bf16):
//   prep:   R[l] = expm(skew(p_l)) (orthogonal -> SVD reorth is identity),
//           alpha[l] = sigmoid(gate + PHI); convert Wi,Wo -> bf16
//   gemm1:  P   = X @ Wi^T + bi                      (f32 out)
//   rot:    ROT = blockdiag-SO8(P) per layer          (bf16 out)
//   gemm2:  RES = ROT @ Wo^T + bo - P                 (bf16 out)
//   gemm3:  OUT = X + alpha_l * (RES @ Wo^T + bo)     (f32 out)
// All GEMMs: v_wmma_f32_16x16x32_bf16, 128x128x32 block tiles, 8 waves.
// ---------------------------------------------------------------------------

typedef __attribute__((ext_vector_type(16))) __bf16 v16bf;
typedef __attribute__((ext_vector_type(2)))  __bf16 v2bf;
typedef __attribute__((ext_vector_type(8)))  float  v8f;

struct alignas(16) U128 { unsigned int x, y, z, w; };
union FragBF { v16bf v; U128 q[2]; };

#define H_DIM   2048
#define L_LAYERS 8
#define M_TOTAL 32768           // L*B*S = 8*2*2048
#define LDSS    40              // padded ushort stride for 32-wide K tiles
#define PHI_F   1.618033988749895f

// Branchless native f32->bf16 (hardware conversion; no EXEC-mask divergence).
__device__ __forceinline__ unsigned int pk2(float a, float b) {
    v2bf p; p.x = (__bf16)a; p.y = (__bf16)b;
    return __builtin_bit_cast(unsigned int, p);
}
__device__ __forceinline__ unsigned short f2bf(float a) {
    __bf16 h = (__bf16)a;
    return __builtin_bit_cast(unsigned short, h);
}

// ---------------------------------------------------------------------------
// Prep: 8x8 expm per layer (scaling & squaring + Taylor) + alpha gates.
// ---------------------------------------------------------------------------
__device__ void mm8(const float a[8][8], const float b[8][8], float c[8][8]) {
    for (int i = 0; i < 8; ++i)
        for (int j = 0; j < 8; ++j) {
            float s = 0.f;
            for (int k = 0; k < 8; ++k) s += a[i][k] * b[k][j];
            c[i][j] = s;
        }
}

__global__ void so8_prep(const float* __restrict__ rp,
                         const float* __restrict__ gates,
                         float* __restrict__ Rout,
                         float* __restrict__ alphaOut) {
    const int t = threadIdx.x;
    if (t >= L_LAYERS) return;
    const int IUa[28] = {0,0,0,0,0,0,0, 1,1,1,1,1,1, 2,2,2,2,2, 3,3,3,3, 4,4,4, 5,5, 6};
    const int JUa[28] = {1,2,3,4,5,6,7, 2,3,4,5,6,7, 3,4,5,6,7, 4,5,6,7, 5,6,7, 6,7, 7};
    float A[8][8], Acc[8][8], T[8][8], Tn[8][8];
    for (int i = 0; i < 8; ++i)
        for (int j = 0; j < 8; ++j) A[i][j] = 0.f;
    float mx = 0.f;
    for (int k = 0; k < 28; ++k) {
        float p = rp[t * 28 + k];
        A[IUa[k]][JUa[k]] =  p;
        A[JUa[k]][IUa[k]] = -p;
        float ap = fabsf(p); if (ap > mx) mx = ap;
    }
    int sf = 0; float m = mx * 8.f;
    while (m > 0.125f && sf < 30) { m *= 0.5f; ++sf; }
    float sc = 1.f; for (int s = 0; s < sf; ++s) sc *= 0.5f;
    for (int i = 0; i < 8; ++i)
        for (int j = 0; j < 8; ++j) A[i][j] *= sc;
    for (int i = 0; i < 8; ++i)
        for (int j = 0; j < 8; ++j) { Acc[i][j] = (i == j) ? 1.f : 0.f; T[i][j] = Acc[i][j]; }
    for (int n = 1; n <= 12; ++n) {
        mm8(T, A, Tn);
        float inv = 1.f / (float)n;
        for (int i = 0; i < 8; ++i)
            for (int j = 0; j < 8; ++j) { T[i][j] = Tn[i][j] * inv; Acc[i][j] += T[i][j]; }
    }
    for (int s = 0; s < sf; ++s) {
        mm8(Acc, Acc, Tn);
        for (int i = 0; i < 8; ++i)
            for (int j = 0; j < 8; ++j) Acc[i][j] = Tn[i][j];
    }
    for (int i = 0; i < 8; ++i)
        for (int j = 0; j < 8; ++j) Rout[t * 64 + i * 8 + j] = Acc[i][j];
    alphaOut[t] = 1.f / (1.f + expf(-(gates[t] + PHI_F)));
}

// ---------------------------------------------------------------------------
// Weight f32 -> bf16 (packed 2 per dword)
// ---------------------------------------------------------------------------
__global__ void cvt_f32_bf16(const float* __restrict__ src,
                             unsigned int* __restrict__ dst, int n4) {
    int i = blockIdx.x * blockDim.x + threadIdx.x;
    if (i >= n4) return;
    float4 f = reinterpret_cast<const float4*>(src)[i];
    dst[2 * i + 0] = pk2(f.x, f.y);
    dst[2 * i + 1] = pk2(f.z, f.w);
}

// ---------------------------------------------------------------------------
// Block-diagonal SO(8) rotation: one thread per 8-chunk. rot_j = sum_i v_i R[j,i]
// ---------------------------------------------------------------------------
__global__ void so8_apply(const float* __restrict__ P,
                          const float* __restrict__ R,
                          unsigned short* __restrict__ Rot) {
    __shared__ float sR[L_LAYERS * 64];
    for (int i = threadIdx.x; i < L_LAYERS * 64; i += blockDim.x) sR[i] = R[i];
    __syncthreads();
    size_t c = (size_t)blockIdx.x * blockDim.x + threadIdx.x;   // 8,388,608 chunks
    size_t row = c >> 8;
    int g = (int)(c & 255);
    int l = (int)(row >> 12);
    const float* p = P + row * H_DIM + g * 8;
    float4 a0 = *reinterpret_cast<const float4*>(p);
    float4 a1 = *reinterpret_cast<const float4*>(p + 4);
    float v[8] = {a0.x, a0.y, a0.z, a0.w, a1.x, a1.y, a1.z, a1.w};
    const float* Rl = sR + l * 64;
    float r[8];
#pragma unroll
    for (int j = 0; j < 8; ++j) {
        float s = 0.f;
#pragma unroll
        for (int i = 0; i < 8; ++i) s += v[i] * Rl[j * 8 + i];
        r[j] = s;
    }
    U128 o;
    o.x = pk2(r[0], r[1]); o.y = pk2(r[2], r[3]);
    o.z = pk2(r[4], r[5]); o.w = pk2(r[6], r[7]);
    *reinterpret_cast<U128*>(Rot + row * H_DIM + g * 8) = o;
}

// ---------------------------------------------------------------------------
// WMMA bf16 GEMM:  C[m,n] = sum_k A[m,k] * W[n,k]  (+ epilogue per MODE)
// MODE 1: A=f32 X,   out Pout  = C + bias                      (f32)
// MODE 2: A=bf16 ROT, out BfOut = bf16(C + bias - Pf32)        (bf16)
// MODE 3: A=bf16 RES, out Fout = X + alpha[l] * (C + bias)     (f32)
// ---------------------------------------------------------------------------
template <int MODE>
__global__ __launch_bounds__(256) void so8_gemm(
    const void* __restrict__ Asrc,
    const unsigned short* __restrict__ Wbf,
    const float* __restrict__ bias,
    const float* __restrict__ Pf32,
    const float* __restrict__ Xorig,
    const float* __restrict__ alpha,
    float* __restrict__ Pout,
    unsigned short* __restrict__ BfOut,
    float* __restrict__ Fout) {
    __shared__ alignas(16) unsigned short ldsA[128 * LDSS];
    __shared__ alignas(16) unsigned short ldsB[128 * LDSS];

    const int tid = threadIdx.x;
    const int m0 = blockIdx.y * 128;
    const int n0 = blockIdx.x * 128;
    const int lane = tid & 31;
    const int wm = ((tid >> 5) & 3) * 32;     // wave M offset (4 waves over M)
    const int wn = ((tid >> 5) >> 2) * 64;    // wave N offset (2 waves over N)
    const int hlf = lane >> 4;                // 0: lanes 0-15, 1: lanes 16-31
    const int l16 = lane & 15;

    const int lrow = tid >> 1;                // 0..127 tile row loaded by this thread
    const int lch = tid & 1;                  // which 16-wide K half

    v8f zero8 = {0.f, 0.f, 0.f, 0.f, 0.f, 0.f, 0.f, 0.f};
    v8f acc[2][4];
#pragma unroll
    for (int i = 0; i < 2; ++i)
#pragma unroll
        for (int j = 0; j < 4; ++j) acc[i][j] = zero8;

    for (int k0 = 0; k0 < H_DIM; k0 += 32) {
        // ---- stage A tile (128 x 32) into LDS as bf16 ----
        if (MODE == 1) {
            const float* g = (const float*)Asrc + (size_t)(m0 + lrow) * H_DIM + k0 + lch * 16;
            float4 f0 = *reinterpret_cast<const float4*>(g + 0);
            float4 f1 = *reinterpret_cast<const float4*>(g + 4);
            float4 f2 = *reinterpret_cast<const float4*>(g + 8);
            float4 f3 = *reinterpret_cast<const float4*>(g + 12);
            U128 u0, u1;
            u0.x = pk2(f0.x, f0.y); u0.y = pk2(f0.z, f0.w);
            u0.z = pk2(f1.x, f1.y); u0.w = pk2(f1.z, f1.w);
            u1.x = pk2(f2.x, f2.y); u1.y = pk2(f2.z, f2.w);
            u1.z = pk2(f3.x, f3.y); u1.w = pk2(f3.z, f3.w);
            *reinterpret_cast<U128*>(&ldsA[lrow * LDSS + lch * 16 + 0]) = u0;
            *reinterpret_cast<U128*>(&ldsA[lrow * LDSS + lch * 16 + 8]) = u1;
            if (k0 + 32 < H_DIM) __builtin_prefetch(g + 32, 0, 1);
        } else {
            const unsigned short* g =
                (const unsigned short*)Asrc + (size_t)(m0 + lrow) * H_DIM + k0 + lch * 16;
            *reinterpret_cast<U128*>(&ldsA[lrow * LDSS + lch * 16 + 0]) =
                *reinterpret_cast<const U128*>(g + 0);
            *reinterpret_cast<U128*>(&ldsA[lrow * LDSS + lch * 16 + 8]) =
                *reinterpret_cast<const U128*>(g + 8);
            if (k0 + 32 < H_DIM) __builtin_prefetch(g + 32, 0, 1);
        }
        // ---- stage B tile: rows n of W (bf16), B[k][n] = W[n][k] ----
        {
            const unsigned short* gw = Wbf + (size_t)(n0 + lrow) * H_DIM + k0 + lch * 16;
            *reinterpret_cast<U128*>(&ldsB[lrow * LDSS + lch * 16 + 0]) =
                *reinterpret_cast<const U128*>(gw + 0);
            *reinterpret_cast<U128*>(&ldsB[lrow * LDSS + lch * 16 + 8]) =
                *reinterpret_cast<const U128*>(gw + 8);
            if (k0 + 32 < H_DIM) __builtin_prefetch(gw + 32, 0, 1);
        }
        __syncthreads();

        // ---- build fragments per ISA VGPR layouts and issue WMMAs ----
        FragBF a[2], b[4];
#pragma unroll
        for (int i = 0; i < 2; ++i) {
            // A 16x32 bf16: lanes<16 hold K{0..7,16..23}; lanes>=16 hold K{8..15,24..31}
            int off = (wm + i * 16 + l16) * LDSS + hlf * 8;
            a[i].q[0] = *reinterpret_cast<const U128*>(&ldsA[off + 0]);
            a[i].q[1] = *reinterpret_cast<const U128*>(&ldsA[off + 16]);
        }
#pragma unroll
        for (int j = 0; j < 4; ++j) {
            // B 32x16 bf16: lanes<16 hold K{0..7,16..23} of col n; lanes>=16 hold K{8..15,24..31}
            int off = (wn + j * 16 + l16) * LDSS + hlf * 8;
            b[j].q[0] = *reinterpret_cast<const U128*>(&ldsB[off + 0]);
            b[j].q[1] = *reinterpret_cast<const U128*>(&ldsB[off + 16]);
        }
#pragma unroll
        for (int i = 0; i < 2; ++i)
#pragma unroll
            for (int j = 0; j < 4; ++j)
                acc[i][j] = __builtin_amdgcn_wmma_f32_16x16x32_bf16(
                    false, a[i].v, false, b[j].v, (short)0, acc[i][j], false, false);
        __syncthreads();
    }

    // ---- epilogue: C layout = VGPR r -> row r + 8*(lane>=16), col lane%16 ----
#pragma unroll
    for (int i = 0; i < 2; ++i)
#pragma unroll
        for (int j = 0; j < 4; ++j) {
            const int nn = n0 + wn + j * 16 + l16;
            const float bn = bias[nn];
#pragma unroll
            for (int r = 0; r < 8; ++r) {
                const int mm = m0 + wm + i * 16 + hlf * 8 + r;
                const size_t idx = (size_t)mm * H_DIM + nn;
                const float v = acc[i][j][r] + bn;
                if (MODE == 1) {
                    Pout[idx] = v;
                } else if (MODE == 2) {
                    BfOut[idx] = f2bf(v - Pf32[idx]);
                } else {
                    Fout[idx] = Xorig[idx] + alpha[mm >> 12] * v;
                }
            }
        }
}

// ---------------------------------------------------------------------------
// Launcher
// ---------------------------------------------------------------------------
extern "C" void kernel_launch(void* const* d_in, const int* in_sizes, int n_in,
                              void* d_out, int out_size, void* d_ws, size_t ws_size,
                              hipStream_t stream) {
    (void)in_sizes; (void)n_in; (void)out_size; (void)ws_size;
    const float* X     = (const float*)d_in[0];  // [8,2,2048,2048]
    const float* rp    = (const float*)d_in[1];  // [8,28]
    const float* gates = (const float*)d_in[2];  // [8]
    const float* Wi    = (const float*)d_in[3];  // [2048,2048]
    const float* bi    = (const float*)d_in[4];  // [2048]
    const float* Wo    = (const float*)d_in[5];  // [2048,2048]
    const float* bo    = (const float*)d_in[6];  // [2048]
    float* out = (float*)d_out;

    char* ws = (char*)d_ws;
    const size_t szW   = (size_t)H_DIM * H_DIM * 2;          // 8 MB bf16 weights
    const size_t offWi = 0;
    const size_t offWo = offWi + szW;
    const size_t offR  = offWo + szW;
    const size_t offAl = offR + 4096;
    const size_t offP  = offAl + 4096;
    const size_t szP   = (size_t)M_TOTAL * H_DIM * 4;        // 268 MB f32
    const size_t offRt = offP + szP;
    const size_t szBf  = (size_t)M_TOTAL * H_DIM * 2;        // 134 MB bf16
    const size_t offRs = offRt + szBf;

    unsigned short* WiBf  = (unsigned short*)(ws + offWi);
    unsigned short* WoBf  = (unsigned short*)(ws + offWo);
    float*          Rws   = (float*)(ws + offR);
    float*          alphW = (float*)(ws + offAl);
    float*          Pws   = (float*)(ws + offP);
    unsigned short* RotBf = (unsigned short*)(ws + offRt);
    unsigned short* ResBf = (unsigned short*)(ws + offRs);

    so8_prep<<<1, 32, 0, stream>>>(rp, gates, Rws, alphW);

    const int n4 = (H_DIM * H_DIM) / 4;                      // 1,048,576 float4s
    cvt_f32_bf16<<<n4 / 256, 256, 0, stream>>>(Wi, (unsigned int*)WiBf, n4);
    cvt_f32_bf16<<<n4 / 256, 256, 0, stream>>>(Wo, (unsigned int*)WoBf, n4);

    dim3 grid(H_DIM / 128, M_TOTAL / 128);                   // (16, 256)
    so8_gemm<1><<<grid, 256, 0, stream>>>(X, WiBf, bi, nullptr, nullptr, nullptr,
                                          Pws, nullptr, nullptr);

    so8_apply<<<(M_TOTAL * (H_DIM / 8)) / 256, 256, 0, stream>>>(Pws, Rws, RotBf);

    so8_gemm<2><<<grid, 256, 0, stream>>>(RotBf, WoBf, bo, Pws, nullptr, nullptr,
                                          nullptr, ResBf, nullptr);

    so8_gemm<3><<<grid, 256, 0, stream>>>(ResBf, WoBf, bo, nullptr, X, alphW,
                                          nullptr, nullptr, out);
}
